// GUNet_3015067041830
// MI455X (gfx1250) — matrix operand order repaired and checked by reading, not compiled, not run on hardware
//
#include <hip/hip_runtime.h>
#include <hip/hip_bf16.h>

// ---------------------------------------------------------------------------
// Graph U-Net forward for MI455X (gfx1250, wave32, WMMA).
//   B=256 graphs, N=256 nodes, F_IN=128, HID=48, DEPTH=4, D_OUT=97 (pad 112)
// Strategy:
//   * augment+gather fused:  Ap = ((A+I)[perm,:]) @ ((A+I)[perm,:])^T, diag=0
//     -> bf16 WMMA 16x16x32, register-blocked 2x2 tiles (A entries small ints:
//        bf16 exact)
//   * GCN: An@(xW)+b == dinv ⊙ ((A+2I) @ (dinv ⊙ (x@W))) + b
//     -> fp32 WMMA 16x16x4, register-blocked MTxNT tiles (NT spans full N)
//   * top-k via stable rank counting (matches jax.lax.top_k ties)
// ---------------------------------------------------------------------------

typedef __attribute__((ext_vector_type(2)))  float  v2f;
typedef __attribute__((ext_vector_type(8)))  float  v8f;
typedef __attribute__((ext_vector_type(16))) __bf16 v16bf;

#define BATCH 256
#define N0    256
#define FIN   128
#define HID   48
#define DOUT  97
#define DPAD  112
#define KSORT 30
#define NC1   16
#define NC2   32
#define KW2   5
#define PLEN  15
#define CLEN  11
#define DENSE 352
#define ODIM  32

static __device__ __forceinline__ float relu_f(float v) { return v > 0.0f ? v : 0.0f; }

// ---------------------------------------------------------------------------
// Register-blocked fp32 WMMA GEMM: C[z] = A[z](MxK) @ B[z](KxN), row-major.
// One wave computes an (MT*16) x (NT*16) block of C.  Requirements:
//   K % 4 == 0, M % (16*MT) == 0, N % (16*NT) == 0  (launch geometry exact).
// f32 fragment layout (wave32):
//   A 16x4 : lane L -> row M=L%16 ; elems {K0,K0+1}, K0 = (L<16?0:2)
//   B 4x16 : lane L -> col N=L%16 ; elems {K0,K0+1}
//   C 16x16: elem v -> (M = v + (L<16?0:8), N = L%16)
// ---------------------------------------------------------------------------
template <int MT, int NT>
__global__ void __launch_bounds__(32)
k_gemm_f32_t(const float* __restrict__ A, const float* __restrict__ Bm,
             float* __restrict__ C, int M, int K, int N,
             long long sA, long long sB, long long sC) {
  const float* Ab = A  + (long long)blockIdx.z * sA;
  const float* Bb = Bm + (long long)blockIdx.z * sB;
  float*       Cb = C  + (long long)blockIdx.z * sC;
  const int lane = threadIdx.x & 31;
  const int half = lane >> 4;
  const int l16  = lane & 15;
  const int mbase = blockIdx.x * (16 * MT);
  const int nbase = blockIdx.y * (16 * NT);

  v8f acc[MT][NT];
  #pragma unroll
  for (int mi = 0; mi < MT; ++mi)
    #pragma unroll
    for (int ni = 0; ni < NT; ++ni)
      acc[mi][ni] = (v8f){};

#if __has_builtin(__builtin_amdgcn_wmma_f32_16x16x4_f32)
  for (int k = 0; k < K; k += 4) {
    const int ka = k + half * 2;
    v2f a[MT], b[NT];
    #pragma unroll
    for (int mi = 0; mi < MT; ++mi) {
      const long long r = (long long)(mbase + mi * 16 + l16) * K + ka;
      a[mi][0] = Ab[r];
      a[mi][1] = Ab[r + 1];
    }
    #pragma unroll
    for (int ni = 0; ni < NT; ++ni) {
      const long long c0 = (long long)ka * N + nbase + ni * 16 + l16;
      b[ni][0] = Bb[c0];
      b[ni][1] = Bb[c0 + N];
    }
    if (k + 16 < K) {
      // pull the next K-slab toward the WGP while the matrix pipe drains
      __builtin_prefetch(&Ab[(long long)(mbase + l16) * K + k + 16], 0, 1);
      __builtin_prefetch(&Bb[(long long)(k + 16) * N + nbase + l16], 0, 1);
    }
    #pragma unroll
    for (int mi = 0; mi < MT; ++mi)
      #pragma unroll
      for (int ni = 0; ni < NT; ++ni)
        acc[mi][ni] = __builtin_amdgcn_wmma_f32_16x16x4_f32(
            false, a[mi], false, b[ni], (short)0, acc[mi][ni], false, false);
  }
#else
  for (int k = 0; k < K; ++k)
    #pragma unroll
    for (int mi = 0; mi < MT; ++mi)
      #pragma unroll
      for (int ni = 0; ni < NT; ++ni)
        #pragma unroll
        for (int v = 0; v < 8; ++v)
          acc[mi][ni][v] += Ab[(long long)(mbase + mi * 16 + half * 8 + v) * K + k] *
                            Bb[(long long)k * N + nbase + ni * 16 + l16];
#endif

  #pragma unroll
  for (int mi = 0; mi < MT; ++mi) {
    const int r0 = mbase + mi * 16 + half * 8;
    #pragma unroll
    for (int ni = 0; ni < NT; ++ni) {
      const int c = nbase + ni * 16 + l16;
      #pragma unroll
      for (int v = 0; v < 8; ++v)
        Cb[(long long)(r0 + v) * N + c] = acc[mi][ni][v];
    }
  }
}

// ---------------------------------------------------------------------------
// Register-blocked bf16 WMMA: Ap[z] (m x m) = G[z] (m x n, bf16) @ G[z]^T,
// zero diagonal.  One wave computes (MT*16) x (NT*16) of Ap.
// bf16 A-frag 16x32: lane L, row=L%16, elem e -> v=e/2,h=e&1,
//   K = (v>=4?16:0) + (L<16?0:8) + 2*(v&3) + h
// bf16 B-frag 32x16 (from G transposed): lane L, col=L%16,
//   elem e -> K = (e>=8?16:0) + (L<16?0:8) + (e&7)
// ---------------------------------------------------------------------------
template <int MT, int NT>
__global__ void __launch_bounds__(32)
k_apgemm_bf16_t(const __bf16* __restrict__ G, float* __restrict__ Ap,
                int m, int n) {
  const __bf16* Gb  = G  + (long long)blockIdx.z * m * n;
  float*        Apb = Ap + (long long)blockIdx.z * m * m;
  const int lane = threadIdx.x & 31;
  const int half = lane >> 4;
  const int l16  = lane & 15;
  const int mbase = blockIdx.x * (16 * MT);
  const int nbase = blockIdx.y * (16 * NT);

  v8f acc[MT][NT];
  #pragma unroll
  for (int mi = 0; mi < MT; ++mi)
    #pragma unroll
    for (int ni = 0; ni < NT; ++ni)
      acc[mi][ni] = (v8f){};

  for (int k0 = 0; k0 < n; k0 += 32) {
    v16bf a[MT], b[NT];
    #pragma unroll
    for (int mi = 0; mi < MT; ++mi)
      #pragma unroll
      for (int e = 0; e < 16; ++e) {
        const int v = e >> 1, h = e & 1;
        const int ka = k0 + ((v >= 4) ? 16 : 0) + half * 8 + 2 * (v & 3) + h;
        a[mi][e] = Gb[(long long)(mbase + mi * 16 + l16) * n + ka];
      }
    #pragma unroll
    for (int ni = 0; ni < NT; ++ni)
      #pragma unroll
      for (int e = 0; e < 16; ++e) {
        const int kb = k0 + ((e >= 8) ? 16 : 0) + half * 8 + (e & 7);
        b[ni][e] = Gb[(long long)(nbase + ni * 16 + l16) * n + kb];
      }
    #pragma unroll
    for (int mi = 0; mi < MT; ++mi)
      #pragma unroll
      for (int ni = 0; ni < NT; ++ni)
        acc[mi][ni] = __builtin_amdgcn_wmma_f32_16x16x32_bf16(
            false, a[mi], false, b[ni], (short)0, acc[mi][ni], false, false);
  }

  #pragma unroll
  for (int mi = 0; mi < MT; ++mi) {
    const int r0 = mbase + mi * 16 + half * 8;
    #pragma unroll
    for (int ni = 0; ni < NT; ++ni) {
      const int c = nbase + ni * 16 + l16;
      #pragma unroll
      for (int v = 0; v < 8; ++v) {
        const int r = r0 + v;
        Apb[(long long)r * m + c] = (r == c) ? 0.0f : acc[mi][ni][v];
      }
    }
  }
}

// G[b,t,k] = bf16( A[b, perm[b,t], k] + (perm[b,t]==k) )
__global__ void k_stage_gbf(const float* __restrict__ A, const int* __restrict__ perm,
                            __bf16* __restrict__ G, int m, int n, long long total) {
  long long idx = (long long)blockIdx.x * blockDim.x + threadIdx.x;
  if (idx >= total) return;
  const int k = (int)(idx % n);
  const int t = (int)((idx / n) % m);
  const int b = (int)(idx / ((long long)n * m));
  const int p = perm[b * m + t];
  float v = A[((long long)b * n + p) * n + k] + (p == k ? 1.0f : 0.0f);
  G[idx] = (__bf16)v;
}

// dinv[b,r] = rsqrt( 2 + sum_j A[b,r,j] )
__global__ void k_dinv(const float* __restrict__ A, float* __restrict__ dinv, int n) {
  const int b = blockIdx.y;
  const int r = blockIdx.x * blockDim.x + threadIdx.x;
  if (r >= n) return;
  const float* row = A + ((long long)b * n + r) * n;
  float s = 2.0f;
  for (int j = 0; j < n; ++j) s += row[j];
  dinv[b * n + r] = rsqrtf(s);
}

// xw[b,r,c] *= dinv[b,r]
__global__ void k_scale_rows(float* __restrict__ xw, const float* __restrict__ dinv,
                             int n, int ncols, long long total) {
  long long idx = (long long)blockIdx.x * blockDim.x + threadIdx.x;
  if (idx >= total) return;
  const int r = (int)((idx / ncols) % n);
  const int b = (int)(idx / ((long long)n * ncols));
  xw[idx] *= dinv[b * n + r];
}

// h = act( dinv * (y + 2*xs) + bias )   (bias only for c<nbias, pad cols -> 0)
__global__ void k_gcn_epi(const float* __restrict__ y, const float* __restrict__ xs,
                          const float* __restrict__ dinv, const float* __restrict__ bias,
                          float* __restrict__ h, int n, int ncols, int nbias,
                          int dorelu, long long total) {
  long long idx = (long long)blockIdx.x * blockDim.x + threadIdx.x;
  if (idx >= total) return;
  const int c = (int)(idx % ncols);
  const int r = (int)((idx / ncols) % n);
  const int b = (int)(idx / ((long long)n * ncols));
  if (c >= nbias) { h[idx] = 0.0f; return; }
  float v = dinv[b * n + r] * (y[idx] + 2.0f * xs[idx]) + bias[c];
  h[idx] = dorelu ? relu_f(v) : v;
}

// score[b,v] = tanh( dot(h[b,v,:],p) / ||p|| )
__global__ void k_score(const float* __restrict__ h, const float* __restrict__ p,
                        float* __restrict__ sc, int n) {
  const int b = blockIdx.y;
  const int v = blockIdx.x * blockDim.x + threadIdx.x;
  if (v >= n) return;
  float pp = 0.0f, d = 0.0f;
  for (int c = 0; c < HID; ++c) {
    const float pc = p[c];
    pp += pc * pc;
    d  += h[((long long)b * n + v) * HID + c] * pc;
  }
  sc[b * n + v] = tanhf(d * rsqrtf(pp));
}

// stable descending top-k by rank counting (ties -> lower index first)
__global__ void k_topk(const float* __restrict__ sc, int n, int k,
                       int* __restrict__ perm, float* __restrict__ vals) {
  __shared__ float s[N0];
  const int b = blockIdx.x;
  for (int i = threadIdx.x; i < n; i += blockDim.x) s[i] = sc[b * n + i];
  __syncthreads();
  for (int i = threadIdx.x; i < n; i += blockDim.x) {
    const float si = s[i];
    int r = 0;
    for (int j = 0; j < n; ++j) {
      const float sj = s[j];
      r += (sj > si) || (sj == si && j < i);
    }
    if (r < k) { perm[b * k + r] = i; vals[b * k + r] = si; }
  }
}

// xp[b,t,:] = h[b,perm[t],:] * vals[b,t]
__global__ void k_gather_xp(const float* __restrict__ h, const int* __restrict__ perm,
                            const float* __restrict__ vals, float* __restrict__ xp,
                            int n, int m, long long total) {
  long long idx = (long long)blockIdx.x * blockDim.x + threadIdx.x;
  if (idx >= total) return;
  const int c = (int)(idx % HID);
  const int t = (int)((idx / HID) % m);
  const int b = (int)(idx / ((long long)HID * m));
  const int p = perm[b * m + t];
  xp[idx] = h[((long long)b * n + p) * HID + c] * vals[b * m + t];
}

__global__ void k_copy(float* __restrict__ dst, const float* __restrict__ src,
                       long long total) {
  long long idx = (long long)blockIdx.x * blockDim.x + threadIdx.x;
  if (idx < total) dst[idx] = src[idx];
}

// dst[b,perm[t],:] += hs[b,t,:]
__global__ void k_scatter_add(float* __restrict__ dst, const float* __restrict__ hs,
                              const int* __restrict__ perm, int n, int m,
                              long long total) {
  long long idx = (long long)blockIdx.x * blockDim.x + threadIdx.x;
  if (idx >= total) return;
  const int c = (int)(idx % HID);
  const int t = (int)((idx / HID) % m);
  const int b = (int)(idx / ((long long)HID * m));
  const int p = perm[b * m + t];
  dst[((long long)b * n + p) * HID + c] += hs[idx];
}

// zero-pad uWl (48x97) into (48x112)
__global__ void k_pad_w(const float* __restrict__ W, float* __restrict__ Wp) {
  int idx = blockIdx.x * blockDim.x + threadIdx.x;
  if (idx >= HID * DPAD) return;
  const int k = idx / DPAD, c = idx % DPAD;
  Wp[idx] = (c < DOUT) ? W[k * DOUT + c] : 0.0f;
}

__global__ void k_lastcol(const float* __restrict__ h97, float* __restrict__ sc,
                          long long total) {
  long long idx = (long long)blockIdx.x * blockDim.x + threadIdx.x;
  if (idx < total) sc[idx] = h97[idx * DPAD + (DOUT - 1)];
}

// sp[b,t,c] = h97[b, order[b,t], c]  (c < 97)
__global__ void k_gather_sp(const float* __restrict__ h97, const int* __restrict__ ord,
                            float* __restrict__ sp, long long total) {
  long long idx = (long long)blockIdx.x * blockDim.x + threadIdx.x;
  if (idx >= total) return;
  const int c = (int)(idx % DOUT);
  const int t = (int)((idx / DOUT) % KSORT);
  const int b = (int)(idx / ((long long)DOUT * KSORT));
  sp[idx] = h97[((long long)b * N0 + ord[b * KSORT + t]) * DPAD + c];
}

// y1[b,c,k] = relu( sum_d sp[b,k,d]*c1W[c,d] + c1b[c] )
__global__ void k_conv1(const float* __restrict__ sp, const float* __restrict__ W,
                        const float* __restrict__ bias, float* __restrict__ y1) {
  int idx = blockIdx.x * blockDim.x + threadIdx.x;
  if (idx >= BATCH * NC1 * KSORT) return;
  const int k = idx % KSORT;
  const int c = (idx / KSORT) % NC1;
  const int b = idx / (KSORT * NC1);
  float s = bias[c];
  for (int d = 0; d < DOUT; ++d)
    s += sp[((long long)b * KSORT + k) * DOUT + d] * W[c * DOUT + d];
  y1[idx] = relu_f(s);
}

__global__ void k_maxpool(const float* __restrict__ y1, float* __restrict__ y2) {
  int idx = blockIdx.x * blockDim.x + threadIdx.x;
  if (idx >= BATCH * NC1 * PLEN) return;
  const int t = idx % PLEN;
  const int c = (idx / PLEN) % NC1;
  const int b = idx / (PLEN * NC1);
  const float a = y1[(b * NC1 + c) * KSORT + 2 * t];
  const float d = y1[(b * NC1 + c) * KSORT + 2 * t + 1];
  y2[idx] = a > d ? a : d;
}

// y3[b,o,t] = relu( sum_{c,dk} y2[b,c,t+dk]*c2W[o,c,dk] + c2b[o] )
__global__ void k_conv2(const float* __restrict__ y2, const float* __restrict__ W,
                        const float* __restrict__ bias, float* __restrict__ y3) {
  int idx = blockIdx.x * blockDim.x + threadIdx.x;
  if (idx >= BATCH * NC2 * CLEN) return;
  const int t = idx % CLEN;
  const int o = (idx / CLEN) % NC2;
  const int b = idx / (CLEN * NC2);
  float s = bias[o];
  for (int c = 0; c < NC1; ++c)
    for (int dk = 0; dk < KW2; ++dk)
      s += y2[(b * NC1 + c) * PLEN + t + dk] * W[(o * NC1 + c) * KW2 + dk];
  y3[idx] = relu_f(s);
}

__global__ void k_dense(const float* __restrict__ y3, const float* __restrict__ W,
                        const float* __restrict__ bias, float* __restrict__ out) {
  int idx = blockIdx.x * blockDim.x + threadIdx.x;
  if (idx >= BATCH * ODIM) return;
  const int u = idx % ODIM;
  const int b = idx / ODIM;
  float s = bias[u];
  for (int m = 0; m < DENSE; ++m)
    s += y3[b * DENSE + m] * W[m * ODIM + u];
  out[idx] = relu_f(s);
}

// ---------------------------------------------------------------------------
static inline void* bump(char*& p, size_t bytes) {
  void* r = (void*)p;
  p += (bytes + 255) & ~(size_t)255;
  return r;
}
static inline int cdiv(long long a, int b) { return (int)((a + b - 1) / b); }

extern "C" void kernel_launch(void* const* d_in, const int* in_sizes, int n_in,
                              void* d_out, int out_size, void* d_ws, size_t ws_size,
                              hipStream_t stream) {
  (void)in_sizes; (void)n_in; (void)out_size; (void)ws_size;
  const float* x      = (const float*)d_in[0];
  const float* A_in   = (const float*)d_in[1];
  const float* dW0    = (const float*)d_in[2];
  const float* db0    = (const float*)d_in[3];
  const float* dW     = (const float*)d_in[4];
  const float* db     = (const float*)d_in[5];
  const float* pool_p = (const float*)d_in[6];
  const float* uW     = (const float*)d_in[7];
  const float* ub     = (const float*)d_in[8];
  const float* uWl    = (const float*)d_in[9];
  const float* ubl    = (const float*)d_in[10];
  const float* c1W    = (const float*)d_in[11];
  const float* c1b    = (const float*)d_in[12];
  const float* c2W    = (const float*)d_in[13];
  const float* c2b    = (const float*)d_in[14];
  const float* oW     = (const float*)d_in[15];
  const float* ob     = (const float*)d_in[16];
  float* out = (float*)d_out;

  char* wp = (char*)d_ws;
  float* h0   = (float*)bump(wp, (size_t)BATCH * 256 * HID * 4);
  float* h1   = (float*)bump(wp, (size_t)BATCH * 128 * HID * 4);
  float* h2   = (float*)bump(wp, (size_t)BATCH * 64  * HID * 4);
  float* h3   = (float*)bump(wp, (size_t)BATCH * 32  * HID * 4);
  float* h4   = (float*)bump(wp, (size_t)BATCH * 16  * HID * 4);
  float* Ap1  = (float*)bump(wp, (size_t)BATCH * 128 * 128 * 4);
  float* Ap2  = (float*)bump(wp, (size_t)BATCH * 64  * 64  * 4);
  float* Ap3  = (float*)bump(wp, (size_t)BATCH * 32  * 32  * 4);
  float* Ap4  = (float*)bump(wp, (size_t)BATCH * 16  * 16  * 4);
  __bf16* Gbf = (__bf16*)bump(wp, (size_t)BATCH * 128 * 256 * 2);
  float* xw   = (float*)bump(wp, (size_t)BATCH * 256 * DPAD * 4);
  float* yb   = (float*)bump(wp, (size_t)BATCH * 256 * DPAD * 4);
  float* dinv = (float*)bump(wp, (size_t)BATCH * 256 * 4);
  float* sc   = (float*)bump(wp, (size_t)BATCH * 256 * 4);
  int*  perm0 = (int*)  bump(wp, (size_t)BATCH * 128 * 4);
  int*  perm1 = (int*)  bump(wp, (size_t)BATCH * 64  * 4);
  int*  perm2 = (int*)  bump(wp, (size_t)BATCH * 32  * 4);
  int*  perm3 = (int*)  bump(wp, (size_t)BATCH * 16  * 4);
  float* vals = (float*)bump(wp, (size_t)BATCH * 128 * 4);
  float* xp   = (float*)bump(wp, (size_t)BATCH * 128 * HID * 4);
  float* hua  = (float*)bump(wp, (size_t)BATCH * 256 * HID * 4);
  float* hub  = (float*)bump(wp, (size_t)BATCH * 256 * HID * 4);
  float* h97  = (float*)bump(wp, (size_t)BATCH * 256 * DPAD * 4);
  float* Wpad = (float*)bump(wp, (size_t)HID * DPAD * 4);
  int*  order = (int*)  bump(wp, (size_t)BATCH * KSORT * 4);
  float* sp   = (float*)bump(wp, (size_t)BATCH * KSORT * DOUT * 4);
  float* y1   = (float*)bump(wp, (size_t)BATCH * NC1 * KSORT * 4);
  float* y2   = (float*)bump(wp, (size_t)BATCH * NC1 * PLEN * 4);
  float* y3   = (float*)bump(wp, (size_t)BATCH * NC2 * CLEN * 4);

  float* hlev[5]    = { h0, h1, h2, h3, h4 };
  float* Alev[5]    = { (float*)A_in, Ap1, Ap2, Ap3, Ap4 };
  int*   permlev[4] = { perm0, perm1, perm2, perm3 };

  const int TB = 256;
  auto EW = [&](long long total) { return dim3(cdiv(total, TB)); };

  // ---------- level-0 GCN: h0 = relu(An @ (x@dW0) + db0) ----------
  {
    long long rows = (long long)BATCH * N0;
    k_gemm_f32_t<2, 3><<<dim3((unsigned)(rows / 32), 1, 1), dim3(32), 0, stream>>>(
        x, dW0, xw, (int)rows, FIN, HID, 0LL, 0LL, 0LL);
    k_dinv<<<dim3(cdiv(N0, 128), BATCH), dim3(128), 0, stream>>>(A_in, dinv, N0);
    long long tot = rows * HID;
    k_scale_rows<<<EW(tot), dim3(TB), 0, stream>>>(xw, dinv, N0, HID, tot);
    k_gemm_f32_t<2, 3><<<dim3(N0 / 32, 1, BATCH), dim3(32), 0, stream>>>(
        A_in, xw, yb, N0, N0, HID,
        (long long)N0 * N0, (long long)N0 * HID, (long long)N0 * HID);
    k_gcn_epi<<<EW(tot), dim3(TB), 0, stream>>>(yb, xw, dinv, db0, h0, N0, HID, HID, 1, tot);
  }

  // ---------- down path ----------
  for (int i = 0; i < 4; ++i) {
    const int n = N0 >> i, m = n >> 1;
    k_score<<<dim3(cdiv(n, 128), BATCH), dim3(128), 0, stream>>>(
        hlev[i], pool_p + i * HID, sc, n);
    k_topk<<<dim3(BATCH), dim3(256), 0, stream>>>(sc, n, m, permlev[i], vals);
    long long gt = (long long)BATCH * m * n;
    k_stage_gbf<<<EW(gt), dim3(TB), 0, stream>>>(Alev[i], permlev[i], Gbf, m, n, gt);
    if (m >= 32)
      k_apgemm_bf16_t<2, 2><<<dim3(m / 32, m / 32, BATCH), dim3(32), 0, stream>>>(
          Gbf, Alev[i + 1], m, n);
    else
      k_apgemm_bf16_t<1, 1><<<dim3(1, 1, BATCH), dim3(32), 0, stream>>>(
          Gbf, Alev[i + 1], m, n);
    long long xt = (long long)BATCH * m * HID;
    k_gather_xp<<<EW(xt), dim3(TB), 0, stream>>>(hlev[i], permlev[i], vals, xp, n, m, xt);
    long long rows = (long long)BATCH * m;
    k_gemm_f32_t<2, 3><<<dim3((unsigned)(rows / 32), 1, 1), dim3(32), 0, stream>>>(
        xp, dW + i * HID * HID, xw, (int)rows, HID, HID, 0LL, 0LL, 0LL);
    k_dinv<<<dim3(cdiv(m, 128), BATCH), dim3(128), 0, stream>>>(Alev[i + 1], dinv, m);
    k_scale_rows<<<EW(xt), dim3(TB), 0, stream>>>(xw, dinv, m, HID, xt);
    if (m >= 32)
      k_gemm_f32_t<2, 3><<<dim3(m / 32, 1, BATCH), dim3(32), 0, stream>>>(
          Alev[i + 1], xw, yb, m, m, HID,
          (long long)m * m, (long long)m * HID, (long long)m * HID);
    else
      k_gemm_f32_t<1, 3><<<dim3(1, 1, BATCH), dim3(32), 0, stream>>>(
          Alev[i + 1], xw, yb, m, m, HID,
          (long long)m * m, (long long)m * HID, (long long)m * HID);
    k_gcn_epi<<<EW(xt), dim3(TB), 0, stream>>>(
        yb, xw, dinv, db + i * HID, hlev[i + 1], m, HID, HID, 1, xt);
  }

  // ---------- up path (i = 0..2, HID->HID GCNs) ----------
  float* hcur = h4;
  for (int i = 0; i < 3; ++i) {
    const int j = 3 - i;
    const int n = N0 >> j, m = n >> 1;   // hcur has m nodes
    long long nt = (long long)BATCH * n * HID;
    long long mt = (long long)BATCH * m * HID;
    k_copy<<<EW(nt), dim3(TB), 0, stream>>>(hub, hlev[j], nt);
    k_scatter_add<<<EW(mt), dim3(TB), 0, stream>>>(hub, hcur, permlev[j], n, m, mt);
    long long rows = (long long)BATCH * n;
    k_gemm_f32_t<2, 3><<<dim3((unsigned)(rows / 32), 1, 1), dim3(32), 0, stream>>>(
        hub, uW + i * HID * HID, xw, (int)rows, HID, HID, 0LL, 0LL, 0LL);
    k_dinv<<<dim3(cdiv(n, 128), BATCH), dim3(128), 0, stream>>>(Alev[j], dinv, n);
    k_scale_rows<<<EW(nt), dim3(TB), 0, stream>>>(xw, dinv, n, HID, nt);
    k_gemm_f32_t<2, 3><<<dim3(n / 32, 1, BATCH), dim3(32), 0, stream>>>(
        Alev[j], xw, yb, n, n, HID,
        (long long)n * n, (long long)n * HID, (long long)n * HID);
    k_gcn_epi<<<EW(nt), dim3(TB), 0, stream>>>(
        yb, xw, dinv, ub + i * HID, hua, n, HID, HID, 1, nt);
    hcur = hua;
  }

  // ---------- final up step: HID -> 97 (pad 112), no relu ----------
  {
    long long nt = (long long)BATCH * N0 * HID;
    long long mt = (long long)BATCH * 128 * HID;
    long long pt = (long long)BATCH * N0 * DPAD;
    k_copy<<<EW(nt), dim3(TB), 0, stream>>>(hub, h0, nt);
    k_scatter_add<<<EW(mt), dim3(TB), 0, stream>>>(hub, hcur, perm0, N0, 128, mt);
    k_pad_w<<<dim3(cdiv(HID * DPAD, TB)), dim3(TB), 0, stream>>>(uWl, Wpad);
    long long rows = (long long)BATCH * N0;
    k_gemm_f32_t<2, 7><<<dim3((unsigned)(rows / 32), 1, 1), dim3(32), 0, stream>>>(
        hub, Wpad, xw, (int)rows, HID, DPAD, 0LL, 0LL, 0LL);
    k_dinv<<<dim3(cdiv(N0, 128), BATCH), dim3(128), 0, stream>>>(A_in, dinv, N0);
    k_scale_rows<<<EW(pt), dim3(TB), 0, stream>>>(xw, dinv, N0, DPAD, pt);
    k_gemm_f32_t<2, 7><<<dim3(N0 / 32, 1, BATCH), dim3(32), 0, stream>>>(
        A_in, xw, yb, N0, N0, DPAD,
        (long long)N0 * N0, (long long)N0 * DPAD, (long long)N0 * DPAD);
    k_gcn_epi<<<EW(pt), dim3(TB), 0, stream>>>(
        yb, xw, dinv, ubl, h97, N0, DPAD, DOUT, 0, pt);
  }

  // ---------- sort-pool + conv head ----------
  {
    long long lt = (long long)BATCH * N0;
    k_lastcol<<<EW(lt), dim3(TB), 0, stream>>>(h97, sc, lt);
    k_topk<<<dim3(BATCH), dim3(256), 0, stream>>>(sc, N0, KSORT, order, vals);
    long long st = (long long)BATCH * KSORT * DOUT;
    k_gather_sp<<<EW(st), dim3(TB), 0, stream>>>(h97, order, sp, st);
    k_conv1<<<dim3(cdiv(BATCH * NC1 * KSORT, TB)), dim3(TB), 0, stream>>>(sp, c1W, c1b, y1);
    k_maxpool<<<dim3(cdiv(BATCH * NC1 * PLEN, TB)), dim3(TB), 0, stream>>>(y1, y2);
    k_conv2<<<dim3(cdiv(BATCH * NC2 * CLEN, TB)), dim3(TB), 0, stream>>>(y2, c2W, c2b, y3);
    k_dense<<<dim3(cdiv(BATCH * ODIM, TB)), dim3(TB), 0, stream>>>(y3, oW, ob, out);
  }
}
